// SequenceLTCClassifier_26920855011540
// MI455X (gfx1250) — compile-verified, matrix operand order checked by reading, not run on hardware
//
#include <hip/hip_runtime.h>
#include <hip/hip_bf16.h>
#include <stddef.h>

// ---------------------------------------------------------------------------
// LTC sequence classifier, fused persistent-scan implementation for gfx1250.
//   B=4096, T=IMG=64, input width KIN=3*64=192, HID=256, K_total=448.
// Persistent blocks own 32 batch rows and run the whole 64-step scan with
// register-resident gate fusion: each wave owns one 16-wide n-tile and
// accumulates ff/gate/tau pre-activations in 6 v8f WMMA accumulators.
// K loops are deliberately NOT unrolled so the loop-invariant weight
// fragments are re-streamed from L2 each step instead of being hoisted and
// spilled to scratch by the register allocator.
// ---------------------------------------------------------------------------

#define BATCH   4096
#define IMG     64
#define INCH    3
#define HID     256
#define KIN     192            // 3*64
#define NKCH    14             // (192+256)/32 k-chunks
#define NTILES  16             // 256/16 n-tiles (one per wave)
#define MROWS   32             // batch rows per block
#define NTHREADS 512           // 16 wave32s
#define TAU_MIN 0.5f
#define TAU_RNG 1.5f

typedef __attribute__((ext_vector_type(16))) _Float16 v16h;
typedef __attribute__((ext_vector_type(8)))  _Float16 v8h;
typedef __attribute__((ext_vector_type(8)))  float    v8f;

// Packed B-fragment weight buffer:
//   index = (((g*NKCH + kk)*NTILES + nt)*32 + lane)*16 + j
// CDNA5 16-bit B-matrix (32x16, KxN) VGPR striping:
//   lane < 16 : N = nt*16+lane,    K = kk*32 + (j<8 ? j   : j+8)
//   lane >=16 : N = nt*16+lane-16, K = kk*32 + (j<8 ? j+8 : j+16)
// K < 192 comes from *_in_w[n][k]; K >= 192 from *_rec_w[n][k-192].
#define WPSTEP  (NTILES * 32 * 16)            // halves per k-chunk  = 8192
#define GSTRIDE (NKCH * WPSTEP)               // halves per gate     = 114688
#define WB_TOT  (3 * GSTRIDE)

__global__ void pack_weights(const float* __restrict__ ffin,
                             const float* __restrict__ ffrec,
                             const float* __restrict__ gin,
                             const float* __restrict__ grec,
                             const float* __restrict__ tin,
                             const float* __restrict__ trec,
                             _Float16* __restrict__ Wb) {
  int t = blockIdx.x * blockDim.x + threadIdx.x;
  if (t >= WB_TOT) return;
  int j    = t & 15;
  int u    = t >> 4;
  int lane = u & 31;  u >>= 5;
  int nt   = u & 15;  u >>= 4;
  int kk   = u % NKCH; u /= NKCH;
  int g    = u;                              // 0=ff, 1=gate, 2=tau
  int n    = nt * 16 + (lane & 15);
  int koff = (lane < 16) ? ((j < 8) ? j     : j + 8)
                         : ((j < 8) ? j + 8 : j + 16);
  int k    = kk * 32 + koff;
  const float* inw  = (g == 0) ? ffin  : (g == 1) ? gin  : tin;
  const float* recw = (g == 0) ? ffrec : (g == 1) ? grec : trec;
  float v = (k < KIN) ? inw[n * KIN + k] : recw[n * HID + (k - KIN)];
  Wb[t] = (_Float16)v;
}

__device__ __forceinline__ float fast_sigmoid(float x) {
  return 1.0f / (1.0f + __expf(-x));
}

#define WMMA16(a, b, c) \
  __builtin_amdgcn_wmma_f32_16x16x32_f16(false, (a), false, (b), (short)0, (c), false, false)

__global__ __launch_bounds__(NTHREADS)
void ltc_scan(const float* __restrict__ x,
              const _Float16* __restrict__ Wb,
              const float* __restrict__ ff_in_b, const float* __restrict__ ff_bias,
              const float* __restrict__ g_in_b,  const float* __restrict__ g_bias,
              const float* __restrict__ t_in_b,  const float* __restrict__ t_bias,
              const float* __restrict__ head_w,  const float* __restrict__ head_b,
              float* __restrict__ out) {
  // LDS: Ax [32][192] f16 (x slice), Ah [32][256] f16 (hidden state),
  //      bsum [3][256] f32 (in_b + bias, hoisted).  Total 31744 B.
  extern __shared__ char smem[];
  _Float16* Ax   = (_Float16*)smem;                                  // 12288 B
  _Float16* Ah   = (_Float16*)(smem + MROWS * KIN * 2);              // 16384 B
  float*    bsum = (float*)(smem + MROWS * KIN * 2 + MROWS * HID * 2);

  const int b0    = blockIdx.x * MROWS;
  const int tid   = threadIdx.x;
  const int lane  = tid & 31;
  const int wave  = tid >> 5;        // 0..15  == n-tile index
  const int nt    = wave;
  const int laneN = lane & 15;
  const int hi    = (lane >= 16);
  const int mB    = hi ? 8 : 0;      // C/D row base per lane half
  const int n     = nt * 16 + laneN; // this lane's output column

  // ---- init: h(0)=0, combined biases, stage x slice for t=0 ----
  for (int i = tid; i < MROWS * HID; i += NTHREADS)
    Ah[i] = (_Float16)0.0f;
  for (int i = tid; i < 3 * HID; i += NTHREADS) {
    const float* ib = (i < HID) ? ff_in_b : (i < 2 * HID) ? g_in_b : t_in_b;
    const float* bb = (i < HID) ? ff_bias : (i < 2 * HID) ? g_bias : t_bias;
    bsum[i] = ib[i & 255] + bb[i & 255];
  }
  for (int i = tid; i < MROWS * KIN; i += NTHREADS) {
    int row = i / KIN, k = i - row * KIN;
    int c = k >> 6, w = k & 63;
    Ax[i] = (_Float16)x[(((size_t)(b0 + row) * INCH + c) * IMG + 0) * IMG + w];
  }
  __syncthreads();

  // Per-lane invariant base pointers for the GEMM loops.
  const int off = hi ? 8 : 0;
  const _Float16* axp = Ax + laneN * KIN + off;   // A M-tile 0, x part
  const _Float16* ahp = Ah + laneN * HID + off;   // A M-tile 0, h part
  const _Float16* wp0 = Wb + (((size_t)nt) * 32 + lane) * 16;  // kk = 0

  for (int t = 0; t < IMG; ++t) {
    // ---- GEMM: this wave's n-tile, all 3 gates, both 16-row M-tiles ----
    v8f cf0 = {}, cf1 = {}, cg0 = {}, cg1 = {}, ct0 = {}, ct1 = {};

    // x contribution: kk = 0..5 (K = 0..191)
#pragma unroll 1
    for (int kk = 0; kk < 6; ++kk) {
      union { v16h v; v8h h2[2]; } a0, a1;
      const _Float16* ap = axp + kk * 32;
      a0.h2[0] = *(const v8h*)(ap);
      a0.h2[1] = *(const v8h*)(ap + 16);
      a1.h2[0] = *(const v8h*)(ap + 16 * KIN);
      a1.h2[1] = *(const v8h*)(ap + 16 * KIN + 16);
      const _Float16* w = wp0 + (size_t)kk * WPSTEP;
      v16h bf = *(const v16h*)(w);
      v16h bg = *(const v16h*)(w + GSTRIDE);
      v16h bt = *(const v16h*)(w + 2 * GSTRIDE);
      cf0 = WMMA16(a0.v, bf, cf0);  cf1 = WMMA16(a1.v, bf, cf1);
      cg0 = WMMA16(a0.v, bg, cg0);  cg1 = WMMA16(a1.v, bg, cg1);
      ct0 = WMMA16(a0.v, bt, ct0);  ct1 = WMMA16(a1.v, bt, ct1);
    }
    // h contribution: kk = 6..13 (K = 192..447)
#pragma unroll 1
    for (int kk = 0; kk < 8; ++kk) {
      union { v16h v; v8h h2[2]; } a0, a1;
      const _Float16* ap = ahp + kk * 32;
      a0.h2[0] = *(const v8h*)(ap);
      a0.h2[1] = *(const v8h*)(ap + 16);
      a1.h2[0] = *(const v8h*)(ap + 16 * HID);
      a1.h2[1] = *(const v8h*)(ap + 16 * HID + 16);
      const _Float16* w = wp0 + (size_t)(kk + 6) * WPSTEP;
      v16h bf = *(const v16h*)(w);
      v16h bg = *(const v16h*)(w + GSTRIDE);
      v16h bt = *(const v16h*)(w + 2 * GSTRIDE);
      cf0 = WMMA16(a0.v, bf, cf0);  cf1 = WMMA16(a1.v, bf, cf1);
      cg0 = WMMA16(a0.v, bg, cg0);  cg1 = WMMA16(a1.v, bg, cg1);
      ct0 = WMMA16(a0.v, bt, ct0);  ct1 = WMMA16(a1.v, bt, ct1);
    }
    __syncthreads();   // everyone done reading Ax / Ah

    // ---- register-resident LTC update, write h(t+1) into Ah ----
    {
      const float bf = bsum[n];
      const float bg = bsum[HID + n];
      const float bt = bsum[2 * HID + n];
#pragma unroll
      for (int m = 0; m < 2; ++m) {
        v8f cf = m ? cf1 : cf0;
        v8f cg = m ? cg1 : cg0;
        v8f ct = m ? ct1 : ct0;
        const int rowbase = m * 16 + mB;
#pragma unroll
        for (int r = 0; r < 8; ++r) {
          float cand = tanhf(cf[r] + bf);
          float gate = fast_sigmoid(cg[r] + bg);
          float tau  = TAU_MIN + TAU_RNG * fast_sigmoid(ct[r] + bt);
          int   row  = rowbase + r;
          float h    = (float)Ah[row * HID + n];
          float hn   = tanhf(h + (gate * cand - h) / tau);
          Ah[row * HID + n] = (_Float16)hn;
        }
      }
    }
    // ---- stage x slice for t+1 (hidden behind the update) ----
    if (t + 1 < IMG) {
      for (int i = tid; i < MROWS * KIN; i += NTHREADS) {
        int row = i / KIN, k = i - row * KIN;
        int c = k >> 6, w = k & 63;
        Ax[i] = (_Float16)
            x[(((size_t)(b0 + row) * INCH + c) * IMG + (t + 1)) * IMG + w];
      }
    }
    __syncthreads();
  }

  // ---- head: out[b, o] = h . head_w[o] + head_b[o] ----
  for (int i = tid; i < MROWS * 10; i += NTHREADS) {
    int row = i / 10, o = i - row * 10;
    float acc = head_b[o];
    const _Float16* hp = Ah + row * HID;
    const float*    wp = head_w + o * HID;
#pragma unroll 8
    for (int k = 0; k < HID; ++k) acc += (float)hp[k] * wp[k];
    out[(size_t)(b0 + row) * 10 + o] = acc;
  }
}

extern "C" void kernel_launch(void* const* d_in, const int* in_sizes, int n_in,
                              void* d_out, int out_size, void* d_ws, size_t ws_size,
                              hipStream_t stream) {
  (void)in_sizes; (void)n_in; (void)out_size; (void)ws_size;
  const float* x        = (const float*)d_in[0];
  const float* ff_in_w  = (const float*)d_in[1];
  const float* ff_in_b  = (const float*)d_in[2];
  const float* ff_rec_w = (const float*)d_in[3];
  const float* ff_bias  = (const float*)d_in[4];
  const float* g_in_w   = (const float*)d_in[5];
  const float* g_in_b   = (const float*)d_in[6];
  const float* g_rec_w  = (const float*)d_in[7];
  const float* g_bias   = (const float*)d_in[8];
  const float* t_in_w   = (const float*)d_in[9];
  const float* t_in_b   = (const float*)d_in[10];
  const float* t_rec_w  = (const float*)d_in[11];
  const float* t_bias   = (const float*)d_in[12];
  const float* head_w   = (const float*)d_in[13];
  const float* head_b   = (const float*)d_in[14];

  _Float16* Wb = (_Float16*)d_ws;   // 688128 B packed f16 B-fragments

  pack_weights<<<(WB_TOT + 255) / 256, 256, 0, stream>>>(
      ff_in_w, ff_rec_w, g_in_w, g_rec_w, t_in_w, t_rec_w, Wb);

  size_t smem = (size_t)MROWS * KIN * 2     // Ax (f16)
              + (size_t)MROWS * HID * 2     // Ah (f16)
              + (size_t)3 * HID * 4;        // bsum (f32)
  ltc_scan<<<BATCH / MROWS, NTHREADS, smem, stream>>>(
      x, Wb,
      ff_in_b, ff_bias, g_in_b, g_bias, t_in_b, t_bias,
      head_w, head_b, (float*)d_out);
}
